// CorrEdgePooling_39187281608965
// MI455X (gfx1250) — compile-verified, matrix operand order checked by reading, not run on hardware
//
#include <hip/hip_runtime.h>
#include <hip/hip_bf16.h>
#include <stdint.h>

#define NN 4096
#define EE 16384
#define FF 1024

typedef __attribute__((ext_vector_type(16))) _Float16 v16h;
typedef __attribute__((ext_vector_type(8)))  float    v8f;
typedef __attribute__((ext_vector_type(4)))  unsigned u32x4;
typedef __attribute__((ext_vector_type(8)))  int      i32x8;
typedef __attribute__((ext_vector_type(4)))  int      i32x4;

// ---------------------------------------------------------------------------
// K1: p1[n] = x[n]·W[0,:F], p2[n] = x[n]·W[0,F:2F]  via V_WMMA_F32_16X16X32_F16
// One wave per 16-row stripe; B has cols {0,1} = W halves, cols 2..15 = 0.
// A layout (16-bit 16x32): lane l<16 -> row l, K in {0..7,16..23};
// lane l+16 -> row l, K in {8..15,24..31}. B assumed symmetric (col per lane).
// C layout: VGPR v: lanes0-15 -> (M=v, N=lane); lanes16-31 -> (M=v+8, N=lane-16)
// ---------------------------------------------------------------------------
__global__ void proj_wmma_k(const float* __restrict__ x, const float* __restrict__ W,
                            float* __restrict__ p1, float* __restrict__ p2) {
  const int lane = threadIdx.x & 31;
  const int hi   = lane >> 4;     // 0 or 1: selects K sub-group
  const int l16  = lane & 15;
  const int rowBase = blockIdx.x * 16;
  const float* xr = x + (size_t)(rowBase + l16) * FF + hi * 8;
  const float* wr = W + (size_t)(l16 & 1) * FF + hi * 8;   // valid only for l16<2
  v8f acc = {};
  for (int k0 = 0; k0 < FF; k0 += 32) {
    v16h a, bm;
#pragma unroll
    for (int j = 0; j < 8; ++j) {
      a[j]     = (_Float16)xr[k0 + j];
      a[j + 8] = (_Float16)xr[k0 + 16 + j];
    }
#pragma unroll
    for (int j = 0; j < 16; ++j) bm[j] = (_Float16)0.f;
    if (l16 < 2) {
#pragma unroll
      for (int j = 0; j < 8; ++j) {
        bm[j]     = (_Float16)wr[k0 + j];
        bm[j + 8] = (_Float16)wr[k0 + 16 + j];
      }
    }
    // EXEC is all-ones here (divergence above has reconverged)
    acc = __builtin_amdgcn_wmma_f32_16x16x32_f16(false, a, false, bm,
                                                 (short)0, acc, false, false);
  }
  if (l16 < 2) {
    float* dp = l16 ? p2 : p1;
#pragma unroll
    for (int v = 0; v < 8; ++v) dp[rowBase + v + hi * 8] = acc[v];
  }
}

// ---------------------------------------------------------------------------
// Softmax over incoming edges of each dst (segment max / sum via atomics)
// ---------------------------------------------------------------------------
__device__ __forceinline__ unsigned f2ord(float f) {
  unsigned u = __float_as_uint(f);
  return (u & 0x80000000u) ? ~u : (u | 0x80000000u);
}
__device__ __forceinline__ float ord2f(unsigned u) {
  return (u & 0x80000000u) ? __uint_as_float(u & 0x7FFFFFFFu) : __uint_as_float(~u);
}

__global__ void seg_init_k(unsigned* __restrict__ segmax, float* __restrict__ denom) {
  int n = blockIdx.x * blockDim.x + threadIdx.x;
  if (n < NN) { segmax[n] = 0u; denom[n] = 0.f; }   // 0 == ord(-inf) lower bound
}

__global__ void raw_k(const int* __restrict__ ei, const float* __restrict__ p1,
                      const float* __restrict__ p2, const float* __restrict__ b,
                      float* __restrict__ raw, unsigned* __restrict__ segmax) {
  int e = blockIdx.x * blockDim.x + threadIdx.x;
  if (e >= EE) return;
  int s = ei[e], d = ei[EE + e];
  float r = p1[s] + p2[d] + b[0];
  raw[e] = r;
  atomicMax(&segmax[d], f2ord(r));
}

__global__ void exp_k(const int* __restrict__ ei, const float* __restrict__ raw,
                      const unsigned* __restrict__ segmax,
                      float* __restrict__ exv, float* __restrict__ denom) {
  int e = blockIdx.x * blockDim.x + threadIdx.x;
  if (e >= EE) return;
  int d = ei[EE + e];
  float ex = __expf(raw[e] - ord2f(segmax[d]));
  exv[e] = ex;
  atomicAdd(&denom[d], ex);
}

__global__ void score_k(const int* __restrict__ ei, const float* __restrict__ exv,
                        const float* __restrict__ denom, float* __restrict__ score) {
  int e = blockIdx.x * blockDim.x + threadIdx.x;
  if (e >= EE) return;
  int d = ei[EE + e];
  score[e] = exv[e] / denom[d] + 0.5f;
}

// ---------------------------------------------------------------------------
// K3: single-block bitonic argsort (descending by score) entirely in LDS.
// Emits ordered endpoint arrays + sorted scores for the contraction stage.
// ---------------------------------------------------------------------------
__global__ void __launch_bounds__(1024) sort_k(const float* __restrict__ score,
                                               const int* __restrict__ ei,
                                               int* __restrict__ sord,
                                               int* __restrict__ dord,
                                               float* __restrict__ scord) {
  extern __shared__ char smem[];
  float* kk = (float*)smem;
  int*   vv = (int*)(smem + (size_t)EE * sizeof(float));
  const int tid = threadIdx.x, nt = blockDim.x;
  for (int i = tid; i < EE; i += nt) { kk[i] = score[i]; vv[i] = i; }
  __syncthreads();
  for (int size = 2; size <= EE; size <<= 1) {
    for (int stride = size >> 1; stride > 0; stride >>= 1) {
      for (int i = tid; i < EE / 2; i += nt) {
        int idx = (i / stride) * (stride << 1) + (i % stride);
        int par = idx + stride;
        bool descBlk = ((idx & size) == 0);
        float ka = kk[idx], kb = kk[par];
        if (descBlk ? (ka < kb) : (ka > kb)) {
          kk[idx] = kb; kk[par] = ka;
          int t = vv[idx]; vv[idx] = vv[par]; vv[par] = t;
        }
      }
      __syncthreads();
    }
  }
  for (int i = tid; i < EE; i += nt) {
    int e = vv[i];
    sord[i]  = ei[e];
    dord[i]  = ei[EE + e];
    scord[i] = kk[i];
  }
}

// ---------------------------------------------------------------------------
// TDM 1-D staged load: global -> LDS via tensor_load_to_lds (TENSORcnt).
// D# group0: count=1, lds_addr, global_addr[56:0], type=2.
// D# group1: data_size=4B, tensor_dim0=n, tensor_dim1=1, tile_dim0=n, stride0=n.
// ---------------------------------------------------------------------------
#if __has_builtin(__builtin_amdgcn_tensor_load_to_lds)
#define HAVE_TDM 1
__device__ __forceinline__ void tdm_load_1d(unsigned lds_addr, const void* gptr, unsigned ndw) {
  unsigned long long ga = (unsigned long long)gptr;
  u32x4 g0;
  g0[0] = 1u;                                              // count=1, user mode
  g0[1] = lds_addr;                                        // lds_addr [63:32]
  g0[2] = (unsigned)(ga & 0xFFFFFFFFull);                  // global_addr lo
  g0[3] = (unsigned)((ga >> 32) & 0x1FFFFFFull) | (2u << 30); // addr hi | type=2
  i32x8 g1;
  g1[0] = (int)(2u << 16);                                 // data_size = 4 bytes
  g1[1] = (int)((ndw & 0xFFFFu) << 16);                    // tensor_dim0[15:0] @63:48
  g1[2] = (int)(((ndw >> 16) & 0xFFFFu) | (1u << 16));     // dim0 hi | tensor_dim1=1
  g1[3] = (int)((ndw & 0xFFFFu) << 16);                    // tile_dim0 @127:112
  g1[4] = 0;                                               // tile_dim1/2 unused
  g1[5] = (int)ndw;                                        // tensor_dim0_stride lo
  g1[6] = 0; g1[7] = 0;
  i32x4 gz = {0, 0, 0, 0};
#if defined(__clang_major__) && (__clang_major__ >= 23)
  i32x8 gz8 = {0, 0, 0, 0, 0, 0, 0, 0};
  __builtin_amdgcn_tensor_load_to_lds(g0, g1, gz, gz, gz8, 0);
#else
  __builtin_amdgcn_tensor_load_to_lds(g0, g1, gz, gz, 0);
#endif
}
#else
#define HAVE_TDM 0
#endif

// ---------------------------------------------------------------------------
// K4: greedy contraction. Serial scan re-expressed as rounds:
// `remaining` only changes at selections, so the next selected edge is simply
// the FIRST position >= p with both endpoints remaining under current state.
// All state lives in LDS (edge list 128KB + node state ~20KB < 320KB WGP LDS).
// ---------------------------------------------------------------------------
__global__ void __launch_bounds__(1024) contract_k(
    const int* __restrict__ sord, const int* __restrict__ dord,
    const float* __restrict__ scord,
    int* __restrict__ clusi, float* __restrict__ clusf,
    float* __restrict__ nes, float* __restrict__ numc) {
  extern __shared__ char smem[];
  int* ls    = (int*)smem;            // [EE] ordered src
  int* lt    = ls + EE;               // [EE] ordered dst
  int* lclus = lt + EE;               // [NN]
  int* sums  = lclus + NN;            // [1024] block scan
  int* ctl   = sums + 1024;           // [4] control
  unsigned char* lrem = (unsigned char*)(ctl + 4);   // [NN]
  const int tid = threadIdx.x;
  const int nt  = 1024;

#if HAVE_TDM
  if (tid < 32) {   // wave 0 issues both TDM descriptors, waits on TENSORcnt
    tdm_load_1d((unsigned)(unsigned long long)(void*)ls, sord, EE);
    tdm_load_1d((unsigned)(unsigned long long)(void*)lt, dord, EE);
    __builtin_amdgcn_s_wait_tensorcnt(0);
  }
#else
  for (int e = tid; e < EE; e += nt) { ls[e] = sord[e]; lt[e] = dord[e]; }
#endif
  for (int n = tid; n < NN; n += nt) { lclus[n] = 0; lrem[n] = 1; nes[n] = 1.0f; }
  __syncthreads();

  int p = 0, ic = 0;
  while (p < EE) {
    if (tid == 0) ctl[0] = EE;
    __syncthreads();
    for (int e = p + tid; e < EE; e += nt) {
      if (lrem[ls[e]] && lrem[lt[e]]) { atomicMin(&ctl[0], e); break; }
    }
    __syncthreads();
    int e = ctl[0];
    if (e >= EE) break;
    int s = ls[e], t = lt[e];
    if (tid == 0) nes[ic] = scord[e];          // score of selected edge -> cluster ic
    // delete s, t and every node sharing an edge with s or t (cluster overwrite
    // applies even to already-removed neighbors, matching the reference)
    for (int q = tid; q < EE; q += nt) {
      int a = ls[q], c = lt[q];
      if (a == s || a == t || c == s || c == t) {
        lclus[a] = ic; lclus[c] = ic;
        lrem[a] = 0;  lrem[c] = 0;
      }
    }
    __syncthreads();
    ++ic;
    p = e + 1;
  }
  __syncthreads();

  // rank surviving nodes in ascending node order (block prefix scan)
  const int per = NN / 1024;                   // 4
  int base = tid * per, vals[per], lsum = 0;
#pragma unroll
  for (int j = 0; j < per; ++j) { vals[j] = lrem[base + j]; lsum += vals[j]; }
  sums[tid] = lsum;
  __syncthreads();
  for (int off = 1; off < nt; off <<= 1) {
    int add = (tid >= off) ? sums[tid - off] : 0;
    __syncthreads();
    sums[tid] += add;
    __syncthreads();
  }
  int run = sums[tid] - lsum;                  // exclusive prefix
  int total = sums[nt - 1];
#pragma unroll
  for (int j = 0; j < per; ++j)
    if (vals[j]) { lclus[base + j] = ic + run; ++run; }
  __syncthreads();
  for (int n = tid; n < NN; n += nt) {
    clusi[n] = lclus[n];
    clusf[n] = (float)lclus[n];
  }
  if (tid == 0) numc[0] = (float)(ic + total);
}

// ---------------------------------------------------------------------------
// new_batch / new_x (segment_sum * score) / coalesce
// ---------------------------------------------------------------------------
__global__ void batch_k(const int* __restrict__ batch, const int* __restrict__ clusi,
                        float* __restrict__ nb) {
  int tid = threadIdx.x;
  for (int n = tid; n < NN; n += blockDim.x) nb[n] = 0.f;
  __syncthreads();
  for (int n = tid; n < NN; n += blockDim.x) nb[clusi[n]] = (float)batch[n];
}

__global__ void zero_x_k(float* __restrict__ nx) {
  size_t i = (size_t)blockIdx.x * blockDim.x + threadIdx.x;
  if (i < (size_t)NN * FF) nx[i] = 0.f;
}

__global__ void scatter_x_k(const float* __restrict__ x, const int* __restrict__ clusi,
                            float* __restrict__ nx) {
  size_t i = (size_t)blockIdx.x * blockDim.x + threadIdx.x;
  if (i >= (size_t)NN * FF) return;
  int n = (int)(i / FF), f = (int)(i % FF);
  atomicAdd(&nx[(size_t)clusi[n] * FF + f], x[i]);
}

__global__ void scale_x_k(float* __restrict__ nx, const float* __restrict__ nes) {
  size_t i = (size_t)blockIdx.x * blockDim.x + threadIdx.x;
  if (i >= (size_t)NN * FF) return;
  nx[i] *= nes[i / FF];
}

__global__ void __launch_bounds__(1024) coalesce_k(const int* __restrict__ ei,
                                                   const int* __restrict__ clusi,
                                                   float* __restrict__ nei,
                                                   float* __restrict__ uniq) {
  extern __shared__ char smem[];
  long long* kk = (long long*)smem;    // [EE]
  const int tid = threadIdx.x, nt = blockDim.x;
  for (int e = tid; e < EE; e += nt)
    kk[e] = (long long)clusi[ei[e]] * NN + (long long)clusi[ei[EE + e]];
  __syncthreads();
  for (int size = 2; size <= EE; size <<= 1) {
    for (int stride = size >> 1; stride > 0; stride >>= 1) {
      for (int i = tid; i < EE / 2; i += nt) {
        int idx = (i / stride) * (stride << 1) + (i % stride);
        int par = idx + stride;
        bool asc = ((idx & size) == 0);
        long long ka = kk[idx], kb = kk[par];
        if (asc ? (ka > kb) : (ka < kb)) { kk[idx] = kb; kk[par] = ka; }
      }
      __syncthreads();
    }
  }
  for (int e = tid; e < EE; e += nt) {
    long long ks = kk[e];
    nei[e]      = (float)(ks / NN);
    nei[EE + e] = (float)(ks % NN);
    uniq[e]     = (e == 0 || ks != kk[e - 1]) ? 1.0f : 0.0f;
  }
}

// ---------------------------------------------------------------------------
extern "C" void kernel_launch(void* const* d_in, const int* in_sizes, int n_in,
                              void* d_out, int out_size, void* d_ws, size_t ws_size,
                              hipStream_t stream) {
  (void)in_sizes; (void)n_in; (void)out_size; (void)ws_size;
  const float* x     = (const float*)d_in[0];
  const int*   ei    = (const int*)d_in[1];
  const int*   batch = (const int*)d_in[2];
  const float* W     = (const float*)d_in[3];
  const float* b     = (const float*)d_in[4];

  // workspace layout
  float*    p1     = (float*)d_ws;             // [NN]
  float*    p2     = p1 + NN;                  // [NN]
  float*    raw    = p2 + NN;                  // [EE]
  unsigned* segmax = (unsigned*)(raw + EE);    // [NN]
  float*    denom  = (float*)(segmax + NN);    // [NN]
  float*    exv    = denom + NN;               // [EE]
  float*    score  = exv + EE;                 // [EE]
  int*      sord   = (int*)(score + EE);       // [EE]
  int*      dord   = sord + EE;                // [EE]
  float*    scord  = (float*)(dord + EE);      // [EE]
  int*      clusi  = (int*)(scord + EE);       // [NN]

  // output layout (all stored as float; ints cast to float), return order
  float* out    = (float*)d_out;
  float* o_newx = out;                          // NN*FF
  float* o_nei  = o_newx + (size_t)NN * FF;     // 2*EE
  float* o_uniq = o_nei + 2 * EE;               // EE
  float* o_nb   = o_uniq + EE;                  // NN
  float* o_clus = o_nb + NN;                    // NN
  float* o_nes  = o_clus + NN;                  // NN
  float* o_numc = o_nes + NN;                   // 1

  const size_t sort_lds = (size_t)EE * 8;
  const size_t ctr_lds  = (size_t)EE * 8 + (size_t)NN * 4 + 1024 * 4 + 16 + NN;
  const size_t coa_lds  = (size_t)EE * 8;
  (void)hipFuncSetAttribute((const void*)sort_k,
      hipFuncAttributeMaxDynamicSharedMemorySize, (int)sort_lds);
  (void)hipFuncSetAttribute((const void*)contract_k,
      hipFuncAttributeMaxDynamicSharedMemorySize, (int)ctr_lds);
  (void)hipFuncSetAttribute((const void*)coalesce_k,
      hipFuncAttributeMaxDynamicSharedMemorySize, (int)coa_lds);

  proj_wmma_k<<<NN / 16, 32, 0, stream>>>(x, W, p1, p2);
  seg_init_k<<<NN / 256, 256, 0, stream>>>(segmax, denom);
  raw_k<<<EE / 256, 256, 0, stream>>>(ei, p1, p2, b, raw, segmax);
  exp_k<<<EE / 256, 256, 0, stream>>>(ei, raw, segmax, exv, denom);
  score_k<<<EE / 256, 256, 0, stream>>>(ei, exv, denom, score);
  sort_k<<<1, 1024, sort_lds, stream>>>(score, ei, sord, dord, scord);
  contract_k<<<1, 1024, ctr_lds, stream>>>(sord, dord, scord, clusi, o_clus, o_nes, o_numc);
  batch_k<<<1, 1024, 0, stream>>>(batch, clusi, o_nb);
  const int nblk = (NN * FF) / 256;
  zero_x_k<<<nblk, 256, 0, stream>>>(o_newx);
  scatter_x_k<<<nblk, 256, 0, stream>>>(x, clusi, o_newx);
  scale_x_k<<<nblk, 256, 0, stream>>>(o_newx, o_nes);
  coalesce_k<<<1, 1024, coa_lds, stream>>>(ei, clusi, o_nei, o_uniq);
}
